// RegionNorm_22196390985921
// MI455X (gfx1250) — matrix operand order repaired and checked by reading, not compiled, not run on hardware
//
#include <hip/hip_runtime.h>
#include <hip/hip_bf16.h>
#include <math.h>

// RegionNorm (SEAN) fused forward for gfx1250.
// B=4, C=64, H=W=128, F=6, D=512, NH=128.
//
// Pipeline (all on `stream`):
//  k_middle : middle_mu[b,f,o] = relu(codes·fc_w + fc_b)                (tiny)
//  k_pmat   : P{g,b}[b,f,k9,c] = sum_d convW[c,d,k9]*middle_mu[b,f,d]   (tiny GEMM;
//             factors the two 512-channel convs out of the hot path)
//  k_labels : label map from one-hot segmap (u8)
//  k_stats  : per-(b,c) instance-norm mean / rstd over H,W (incl. noise)
//  k_actv   : SPADE shared conv 3->128 + ReLU, stored f16 transposed
//             [b][h][w][nh] with a zero halo (130x130) for implicit-GEMM conv
//  k_pack   : pre-swizzle sp_{gamma,beta}_w (64x1152) into CDNA5 WMMA
//             A-fragment layout (16-bit A 16x32, contiguous k: lane half h holds k 16h..16h+15)
//  k_main   : implicit-GEMM SPADE convs via v_wmma_f32_16x16x32_f16,
//             fused with P-gather (region conv), instance-norm apply,
//             sigmoid blend -> writes d_out directly.

typedef __attribute__((ext_vector_type(16))) _Float16 v16h;
typedef __attribute__((ext_vector_type(8)))  float    v8f;
typedef __attribute__((ext_vector_type(4)))  float    v4f;

#define Bn 4
#define Cn 64
#define Hn 128
#define Wn 128
#define Fn 6
#define Dn 512
#define NHn 128

// ---------------- workspace layout (bytes, all 256B aligned) ----------------
#define WS_MM     0u                 // middle_mu: 4*6*512 f32      = 49152
#define WS_PG     49152u             // Pg: 4*6*9*64 f32            = 55296
#define WS_PB     104448u            // Pb                          = 55296
#define WS_MEAN   159744u            // 256 f32
#define WS_RSTD   160768u            // 256 f32
#define WS_LBL    161792u            // 65536 u8
#define WS_APACK  227328u            // 2*4*36*32*16 f16            = 294912
#define WS_ACTV   522240u            // 4*130*130*128 f16           = 17305600
// total ~17.8 MB

// ---------------------------------------------------------------------------
__global__ void k_middle(const float* __restrict__ codes,
                         const float* __restrict__ fc_w,
                         const float* __restrict__ fc_b,
                         float* __restrict__ mm) {
  int bf = blockIdx.x;               // b*6+f
  int f  = bf % Fn;
  const float* cv = codes + (size_t)bf * Dn;
  const float* wb = fc_w + (size_t)f * Dn * Dn;
  for (int o = threadIdx.x; o < Dn; o += blockDim.x) {
    const float* wr = wb + (size_t)o * Dn;
    float acc = fc_b[f * Dn + o];
    for (int d = 0; d < Dn; ++d) acc = fmaf(cv[d], wr[d], acc);
    mm[(size_t)bf * Dn + o] = acc > 0.f ? acc : 0.f;
  }
}

// P[which][b][f][k9][c] = sum_d convW[c][d][k9] * middle_mu[b][f][d]
__global__ void k_pmat(const float* __restrict__ mm,
                       const float* __restrict__ wg,
                       const float* __restrict__ wbt,
                       float* __restrict__ Pg, float* __restrict__ Pb) {
  int t = blockIdx.x * blockDim.x + threadIdx.x;    // 2*4*6*9*64 = 27648
  if (t >= 27648) return;
  int c = t & 63;
  int k = (t >> 6) % 9;
  int f = (t / 576) % Fn;
  int b = (t / 3456) % Bn;
  int which = t / 13824;
  const float* w = which ? wbt : wg;                // (C,D,3,3)
  const float* m = mm + (size_t)(b * Fn + f) * Dn;
  float acc = 0.f;
  for (int d = 0; d < Dn; ++d)
    acc = fmaf(m[d], w[(size_t)(c * Dn + d) * 9 + k], acc);
  float* P = which ? Pb : Pg;
  P[(((b * Fn + f) * 9) + k) * Cn + c] = acc;
}

__global__ void k_labels(const float* __restrict__ segmap,
                         unsigned char* __restrict__ lbl) {
  int t = blockIdx.x * blockDim.x + threadIdx.x;    // B*H*W = 65536
  if (t >= Bn * Hn * Wn) return;
  int b = t >> 14, hw = t & 16383;
  int j = 0;
  for (int jj = 0; jj < Fn; ++jj)
    if (segmap[((size_t)b * Fn + jj) * (Hn * Wn) + hw] != 0.f) j = jj; // last wins
  lbl[t] = (unsigned char)j;
}

__global__ __launch_bounds__(256)
void k_stats(const float* __restrict__ x, const float* __restrict__ noise,
             const float* __restrict__ nv, float* __restrict__ meanv,
             float* __restrict__ rstdv) {
  int bc = blockIdx.x;                // b*64+c
  int b = bc >> 6, c = bc & 63;
  float nvc = nv[c];
  const float* xp = x + (size_t)bc * (Hn * Wn);
  const float* np = noise + (size_t)b * (Hn * Wn); // noise[b][w][h]
  float s = 0.f, s2 = 0.f;
  for (int i = threadIdx.x; i < Hn * Wn; i += 256) {
    int h = i >> 7, w = i & 127;
    float v = xp[i] + np[w * Hn + h] * nvc;
    s += v; s2 = fmaf(v, v, s2);
  }
  __shared__ float sh[256], sh2[256];
  sh[threadIdx.x] = s; sh2[threadIdx.x] = s2;
  __syncthreads();
  for (int o = 128; o > 0; o >>= 1) {
    if ((int)threadIdx.x < o) {
      sh[threadIdx.x] += sh[threadIdx.x + o];
      sh2[threadIdx.x] += sh2[threadIdx.x + o];
    }
    __syncthreads();
  }
  if (threadIdx.x == 0) {
    float m = sh[0] * (1.f / 16384.f);
    float var = sh2[0] * (1.f / 16384.f) - m * m;
    meanv[bc] = m;
    rstdv[bc] = __frsqrt_rn(var + 1e-5f);
  }
}

// SPADE shared conv (3->128)+ReLU -> f16 transposed layout with zero halo:
// actvT[b][hh=0..129][ww=0..129][nh], (hh,ww) = (h+1,w+1)
__global__ __launch_bounds__(128)
void k_actv(const float* __restrict__ mask, const float* __restrict__ w,
            const float* __restrict__ bias, _Float16* __restrict__ actvT) {
  int blk = blockIdx.x;               // b*130*130 + hh*130 + ww
  int b = blk / 16900, rem = blk - b * 16900;
  int hh = rem / 130, ww = rem - hh * 130;
  int h = hh - 1, wq = ww - 1;
  int nh = threadIdx.x;
  size_t oi = (((size_t)b * 130 + hh) * 130 + ww) * NHn + nh;
  if ((unsigned)h >= (unsigned)Hn || (unsigned)wq >= (unsigned)Wn) {
    actvT[oi] = (_Float16)0.f;
    return;
  }
  float acc = bias[nh];
  const float* wp = w + (size_t)nh * 27;     // (NH,3,3,3)
  #pragma unroll
  for (int ci = 0; ci < 3; ++ci)
    #pragma unroll
    for (int kh = 0; kh < 3; ++kh)
      #pragma unroll
      for (int kw = 0; kw < 3; ++kw) {
        int hn = h + kh - 1, wn = wq + kw - 1;
        if ((unsigned)hn < (unsigned)Hn && (unsigned)wn < (unsigned)Wn)
          acc = fmaf(mask[((size_t)(b * 3 + ci) * Hn + hn) * Wn + wn],
                     wp[(ci * 3 + kh) * 3 + kw], acc);
      }
  acc = acc > 0.f ? acc : 0.f;
  actvT[oi] = (_Float16)acc;
}

// Pre-swizzle sp_{gamma,beta}_w (C=64 x K=1152, K=(kh*3+kw)*128+nh) into the
// WMMA 16-bit A 16x32 per-lane fragments, contiguous k (lane half h holds k = 16h + i).
// Apack flat index: ((which*4+mt)*36 + kc)*512 + lane*16 + i
__global__ void k_pack(const float* __restrict__ wg,
                       const float* __restrict__ wb,
                       _Float16* __restrict__ Apack) {
  int t = blockIdx.x * blockDim.x + threadIdx.x;   // 147456
  if (t >= 147456) return;
  int i    = t & 15;
  int lane = (t >> 4) & 31;
  int kc   = (t >> 9) % 36;
  int mt   = (t / 18432) & 3;
  int which = t / 73728;
  int m = mt * 16 + (lane & 15);
  // lanes 0-15 hold K = 0..15; lanes 16-31 hold K = 16..31 (matches B fill)
  int Kp = (lane < 16) ? i : i + 16;
  int K = kc * 32 + Kp;
  int k9 = K >> 7, nh = K & 127;
  const float* w = which ? wb : wg;                // (C, NH, 3, 3)
  Apack[t] = (_Float16)w[((size_t)(m * NHn + nh)) * 9 + k9];
}

// ---------------------------------------------------------------------------
// Fused main kernel: one workgroup = 64 channels x 32 pixels (one row h).
// 8 waves: wave = (nt<<2)|mt -> mt picks 16-channel tile, nt picks 16-pixel tile.
__global__ __launch_bounds__(256)
void k_main(const float* __restrict__ x, const float* __restrict__ noise,
            const float* __restrict__ noise_var,
            const float* __restrict__ cg_b, const float* __restrict__ cb_b,
            const float* __restrict__ sg_b, const float* __restrict__ sb_b,
            const float* __restrict__ blend_g, const float* __restrict__ blend_b,
            const float* __restrict__ Pg, const float* __restrict__ Pb,
            const float* __restrict__ meanv, const float* __restrict__ rstdv,
            const unsigned char* __restrict__ lbl,
            const _Float16* __restrict__ Apack,
            const _Float16* __restrict__ actvT,
            float* __restrict__ out) {
  const int tid  = threadIdx.x;
  const int lane = tid & 31;
  const int wave = tid >> 5;
  const int mt = wave & 3;
  const int nt = wave >> 2;
  const int b = blockIdx.z;
  const int h = blockIdx.y;
  const int w0 = blockIdx.x * 32 + nt * 16;
  const int ln = lane & 15;
  const int hi = lane >> 4;            // 0 or 1
  const int wpix = w0 + ln;

  v8f accG = {}; v8f accB = {};
  const v16h* Av = (const v16h*)Apack;
  const int agBase = (mt * 36) * 32 + lane;           // gamma weights
  const int abBase = ((4 + mt) * 36) * 32 + lane;     // beta weights

  // K = 1152 = 9 taps * 128 nh, 36 chunks of 32.  For chunk kc:
  //   tap k9 = kc>>2,  nh window = (kc&3)*32.
  #pragma unroll 4
  for (int kc = 0; kc < 36; ++kc) {
    int k9 = kc >> 2;
    int kh = k9 / 3, kw = k9 - kh * 3;
    int nh0 = ((kc & 3) << 5) + (hi << 4);
    int hh = h + kh;                   // halo coords: (h-1+kh)+1
    int ww = wpix + kw;
    const v16h* bp = (const v16h*)(actvT +
        ((((size_t)b * 130 + hh) * 130 + ww) * NHn + nh0));
    v16h bfrag = *bp;                                   // 2x b128, 32B aligned
    v16h ag = Av[agBase + kc * 32];
    v16h ab = Av[abBase + kc * 32];
    accG = __builtin_amdgcn_wmma_f32_16x16x32_f16(
        false, ag, false, bfrag, (short)0, accG, false, false);
    accB = __builtin_amdgcn_wmma_f32_16x16x32_f16(
        false, ab, false, bfrag, (short)0, accB, false, false);
  }

  // ---- region-conv gather (factored 512-channel convs) ----
  const int cbase = mt * 16 + hi * 8;   // C/D layout: VGPR r -> M = r + 8*hi
  float gsum[8], bsum[8];
  #pragma unroll
  for (int r = 0; r < 8; ++r) {
    gsum[r] = cg_b[cbase + r];
    bsum[r] = cb_b[cbase + r];
  }
  #pragma unroll
  for (int k9 = 0; k9 < 9; ++k9) {
    int kh = k9 / 3, kw = k9 - kh * 3;
    int hn = h + kh - 1, wn = wpix + kw - 1;
    if ((unsigned)hn < (unsigned)Hn && (unsigned)wn < (unsigned)Wn) {
      int f = lbl[((size_t)b * Hn + hn) * Wn + wn];
      const float* pg = Pg + ((((size_t)b * Fn + f) * 9 + k9) * Cn + cbase);
      const float* pb = Pb + ((((size_t)b * Fn + f) * 9 + k9) * Cn + cbase);
      v4f g0 = *(const v4f*)pg;       v4f g1 = *(const v4f*)(pg + 4);
      v4f b0 = *(const v4f*)pb;       v4f b1 = *(const v4f*)(pb + 4);
      #pragma unroll
      for (int r = 0; r < 4; ++r) {
        gsum[r] += g0[r]; gsum[r + 4] += g1[r];
        bsum[r] += b0[r]; bsum[r + 4] += b1[r];
      }
    }
  }

  // ---- blend + instance-norm apply ----
  float ga = 1.f / (1.f + __expf(-blend_g[0]));
  float ba = 1.f / (1.f + __expf(-blend_b[0]));
  float nz = noise[(size_t)b * (Hn * Wn) + wpix * Hn + h];   // noise[b][w][h]
  #pragma unroll
  for (int r = 0; r < 8; ++r) {
    int c = cbase + r;
    float gs = accG[r] + sg_b[c];      // gamma_spade
    float bs = accB[r] + sb_b[c];      // beta_spade
    float gf = ga * gsum[r] + (1.f - ga) * gs;
    float bf = ba * bsum[r] + (1.f - ba) * bs;
    int bc = b * Cn + c;
    size_t xi = ((size_t)bc * Hn + h) * Wn + wpix;
    float xn = x[xi] + nz * noise_var[c];
    float xnorm = (xn - meanv[bc]) * rstdv[bc];
    out[xi] = fmaf(xnorm, 1.f + gf, bf);
  }
}

// ---------------------------------------------------------------------------
extern "C" void kernel_launch(void* const* d_in, const int* in_sizes, int n_in,
                              void* d_out, int out_size, void* d_ws, size_t ws_size,
                              hipStream_t stream) {
  const float* x        = (const float*)d_in[0];
  const float* segmap   = (const float*)d_in[1];
  const float* mask     = (const float*)d_in[2];
  const float* codes    = (const float*)d_in[3];
  const float* noise    = (const float*)d_in[4];
  const float* noise_v  = (const float*)d_in[5];
  const float* fc_w     = (const float*)d_in[6];
  const float* fc_b     = (const float*)d_in[7];
  const float* cg_w     = (const float*)d_in[8];
  const float* cg_b     = (const float*)d_in[9];
  const float* cb_w     = (const float*)d_in[10];
  const float* cb_b     = (const float*)d_in[11];
  const float* ss_w     = (const float*)d_in[12];
  const float* ss_b     = (const float*)d_in[13];
  const float* sg_w     = (const float*)d_in[14];
  const float* sg_b     = (const float*)d_in[15];
  const float* sb_w     = (const float*)d_in[16];
  const float* sb_b     = (const float*)d_in[17];
  const float* blend_g  = (const float*)d_in[18];
  const float* blend_b  = (const float*)d_in[19];
  float* out = (float*)d_out;

  char* ws = (char*)d_ws;
  float*          mm    = (float*)(ws + WS_MM);
  float*          Pg    = (float*)(ws + WS_PG);
  float*          Pb    = (float*)(ws + WS_PB);
  float*          meanv = (float*)(ws + WS_MEAN);
  float*          rstdv = (float*)(ws + WS_RSTD);
  unsigned char*  lbl   = (unsigned char*)(ws + WS_LBL);
  _Float16*       Apack = (_Float16*)(ws + WS_APACK);
  _Float16*       actvT = (_Float16*)(ws + WS_ACTV);

  k_middle<<<Bn * Fn, 256, 0, stream>>>(codes, fc_w, fc_b, mm);
  k_pmat<<<(27648 + 255) / 256, 256, 0, stream>>>(mm, cg_w, cb_w, Pg, Pb);
  k_labels<<<(Bn * Hn * Wn + 255) / 256, 256, 0, stream>>>(segmap, lbl);
  k_stats<<<Bn * Cn, 256, 0, stream>>>(x, noise, noise_v, meanv, rstdv);
  k_actv<<<Bn * 130 * 130, 128, 0, stream>>>(mask, ss_w, ss_b, actvT);
  k_pack<<<(147456 + 255) / 256, 256, 0, stream>>>(sg_w, sb_w, Apack);

  dim3 grid(Wn / 32, Hn, Bn);
  k_main<<<grid, 256, 0, stream>>>(x, noise, noise_v, cg_b, cb_b, sg_b, sb_b,
                                   blend_g, blend_b, Pg, Pb, meanv, rstdv,
                                   lbl, Apack, actvT, out);
  (void)in_sizes; (void)n_in; (void)out_size; (void)ws_size;
}
